// PhysicsInformedBasicLiquidNetworkModel_4217657885125
// MI455X (gfx1250) — matrix operand order repaired and checked by reading, not compiled, not run on hardware
//
#include <hip/hip_runtime.h>
#include <hip/hip_bf16.h>

typedef __bf16 bf16_t;
typedef bf16_t v16bf __attribute__((ext_vector_type(16)));
typedef bf16_t v8bf  __attribute__((ext_vector_type(8)));
typedef float  v8f   __attribute__((ext_vector_type(8)));

#define B_SZ   1024
#define T_SZ   2048
#define I_SZ   8
#define H_SZ   256
#define A_SZ   8
#define DT_C   0.1f
#define EPS_C  1e-5f
#define MTILE  16
#define HB_PAD 8      // 16B row pad -> conflict-free ds_load_b128
#define ZS_PAD 1      // 1-float row pad -> half-wave store conflicts removed
#define THREADS 512   // 16 waves (wave32)

__global__ __launch_bounds__(THREADS)
void liquid_net_persistent_kernel(const float* __restrict__ x,
                                  const float* __restrict__ W_in,
                                  const float* __restrict__ b_in,
                                  const float* __restrict__ tau_param,
                                  const float* __restrict__ W_rec,
                                  const float* __restrict__ g1,
                                  const float* __restrict__ beta1,
                                  const float* __restrict__ g2,
                                  const float* __restrict__ beta2,
                                  const float* __restrict__ head_w,
                                  const float* __restrict__ head_b,
                                  float* __restrict__ out)
{
  __shared__ float zs[MTILE][H_SZ + ZS_PAD];                 // pre-LN activations
  __shared__ float hf[MTILE][H_SZ];                          // f32 master state
  __shared__ __align__(16) bf16_t hb[MTILE][H_SZ + HB_PAD];  // bf16 WMMA copy
  __shared__ __align__(16) bf16_t xsb[MTILE][I_SZ];          // current-step inputs (bf16)

  const int tid  = threadIdx.x;
  const int wave = tid >> 5;
  const int lane = tid & 31;
  const int half = lane >> 4;   // lane group 0-15 vs 16-31
  const int lr   = lane & 15;
  const int b0   = blockIdx.x * MTILE;

  // ---------------- persistent per-wave B operands ----------------
  // Wave `wave` owns output columns n0..n0+15. bf16 B 32x16 layout:
  // lane = N; element i<8 -> K = half*8 + i, i>=8 -> K = 16 + half*8 + (i-8).
  const int n = wave * 16 + lr;
  v16bf Bw[8];
  #pragma unroll
  for (int c = 0; c < 8; ++c) {
    const int kbase = c * 32 + half * 8;
    v16bf tv;
    #pragma unroll
    for (int i = 0; i < 16; ++i)
      tv[i] = (bf16_t)W_rec[(size_t)(kbase + i + (i >= 8 ? 8 : 0)) * H_SZ + n];
    Bw[c] = tv;
  }
  // Extra K=32 chunk fusing the input projection: rows 0..7 = W_in, rest 0.
  // Zero rows guarantee the (finite) duplicated A values for K>=8 contribute 0.
  v16bf Bx = {};
  if (half == 0) {
    #pragma unroll
    for (int i = 0; i < I_SZ; ++i)
      Bx[i] = (bf16_t)W_in[(size_t)i * H_SZ + n];
  }
  const float bn = b_in[n];

  // ---------------- per-row params (wave = batch row), cols = lane + 32*j ----
  float g1v[8], b1v[8], itau[8], g2v[8], b2v[8];
  #pragma unroll
  for (int j = 0; j < 8; ++j) {
    const int cidx = lane + 32 * j;
    g1v[j] = g1[cidx];  b1v[j] = beta1[cidx];
    g2v[j] = g2[cidx];  b2v[j] = beta2[cidx];
    const float tp  = tau_param[cidx];
    const float tau = (tp > 20.0f) ? tp : log1pf(expf(tp));   // softplus
    itau[j] = 1.0f / tau;
  }

  // ---------------- h0 = 0 ----------------
  for (int idx = tid; idx < MTILE * H_SZ; idx += THREADS) {
    const int r = idx >> 8, cc = idx & 255;
    hf[r][cc] = 0.0f;
    hb[r][cc] = (bf16_t)0.0f;
  }

  const float* xptr = nullptr;
  if (tid < MTILE * I_SZ)
    xptr = x + (size_t)(b0 + (tid >> 3)) * T_SZ * I_SZ + (tid & 7);

  // ============================ scan over T ============================
  for (int t = 0; t < T_SZ; ++t) {
    if (tid < MTILE * I_SZ) {
      xsb[tid >> 3][tid & 7] = (bf16_t)xptr[(size_t)t * I_SZ];
      if (t + 1 < T_SZ) __builtin_prefetch(xptr + (size_t)(t + 1) * I_SZ, 0, 0);
    }
    __syncthreads();  // h(t), xsb(t) visible

    // ---- preload ALL A tiles (16x ds_load_b128 issue up front) ----
    // bf16 A 16x32 layout: lane row = lr; elements 0..7 -> K = half*8 + 0..7,
    // elements 8..15 -> K = 16 + half*8 + 0..7.
    const bf16_t* rowp = &hb[lr][0];
    v8bf alo[8], ahi[8];
    #pragma unroll
    for (int c = 0; c < 8; ++c) {
      alo[c] = *(const v8bf*)(rowp + c * 32 + half * 8);
      ahi[c] = *(const v8bf*)(rowp + c * 32 + 16 + half * 8);
    }
    // Uniform (branch-free) x chunk: duplicate the 8 inputs into both halves;
    // every position that is not K<8 on lanes 0-15 multiplies a zero B row.
    v8bf xv = *(const v8bf*)(&xsb[lr][0]);
    v16bf Ax = __builtin_shufflevector(xv, xv,
        0, 1, 2, 3, 4, 5, 6, 7, 0, 1, 2, 3, 4, 5, 6, 7);

    // ---- z_tile = [x | h] @ [W_in ; W_rec] via 9 chained bf16 WMMAs ----
    v8f acc = {};
    acc = __builtin_amdgcn_wmma_f32_16x16x32_bf16(false, Ax, false, Bx,
                                                  (short)0, acc, false, false);
    #pragma unroll
    for (int c = 0; c < 8; ++c) {
      v16bf a = __builtin_shufflevector(alo[c], ahi[c],
          0, 1, 2, 3, 4, 5, 6, 7, 8, 9, 10, 11, 12, 13, 14, 15);
      acc = __builtin_amdgcn_wmma_f32_16x16x32_bf16(false, a, false, Bw[c],
                                                    (short)0, acc, false, false);
    }
    {   // C/D layout: VGPR v -> row half*8+v, col = n
      const int rb = half * 8;
      #pragma unroll
      for (int v = 0; v < 8; ++v)
        zs[rb + v][n] = acc[v] + bn;
    }
    __syncthreads();  // z(t) visible; WMMA reads of hb(t)/xsb(t) done

    // ---- LayerNorm + tanh + liquid update: wave w owns batch row w ----
    float s = 0.0f, sq = 0.0f, zv[8];
    #pragma unroll
    for (int j = 0; j < 8; ++j) {
      const float z = zs[wave][lane + 32 * j];
      zv[j] = z; s += z; sq += z * z;
    }
    #pragma unroll
    for (int o = 16; o >= 1; o >>= 1) {
      s  += __shfl_xor(s,  o, 32);
      sq += __shfl_xor(sq, o, 32);
    }
    const float m  = s * (1.0f / H_SZ);
    const float vr = sq * (1.0f / H_SZ) - m * m;
    const float rs = rsqrtf(vr + EPS_C);
    #pragma unroll
    for (int j = 0; j < 8; ++j) {
      const int cidx = lane + 32 * j;
      const float f = tanhf((zv[j] - m) * rs * g1v[j] + b1v[j]);
      float h = hf[wave][cidx];
      h = h + (-h * itau[j] + f) * DT_C;
      h = fminf(10.0f, fmaxf(-10.0f, h));
      hf[wave][cidx] = h;
      hb[wave][cidx] = (bf16_t)h;
    }
  }

  // ---------------- final LN2 + fused 8-wide head ----------------
  float s = 0.0f, sq = 0.0f, hv[8];
  #pragma unroll
  for (int j = 0; j < 8; ++j) {
    const float h = hf[wave][lane + 32 * j];
    hv[j] = h; s += h; sq += h * h;
  }
  #pragma unroll
  for (int o = 16; o >= 1; o >>= 1) {
    s  += __shfl_xor(s,  o, 32);
    sq += __shfl_xor(sq, o, 32);
  }
  const float m  = s * (1.0f / H_SZ);
  const float vr = sq * (1.0f / H_SZ) - m * m;
  const float rs = rsqrtf(vr + EPS_C);
  float p[A_SZ] = {};
  #pragma unroll
  for (int j = 0; j < 8; ++j) {
    const int cidx = lane + 32 * j;
    const float ln = (hv[j] - m) * rs * g2v[j] + b2v[j];
    const float* hwp = head_w + (size_t)cidx * A_SZ;
    #pragma unroll
    for (int a = 0; a < A_SZ; ++a) p[a] += ln * hwp[a];
  }
  #pragma unroll
  for (int a = 0; a < A_SZ; ++a) {
    #pragma unroll
    for (int o = 16; o >= 1; o >>= 1)
      p[a] += __shfl_xor(p[a], o, 32);
  }
  if (lane == 0) {
    #pragma unroll
    for (int a = 0; a < A_SZ; ++a)
      out[(size_t)(b0 + wave) * A_SZ + a] = p[a] + head_b[a];
  }
}

extern "C" void kernel_launch(void* const* d_in, const int* in_sizes, int n_in,
                              void* d_out, int out_size, void* d_ws, size_t ws_size,
                              hipStream_t stream) {
  (void)in_sizes; (void)n_in; (void)d_ws; (void)ws_size; (void)out_size;
  const float* xp   = (const float*)d_in[0];
  const float* Win  = (const float*)d_in[1];
  const float* bin  = (const float*)d_in[2];
  const float* taup = (const float*)d_in[3];
  const float* Wrec = (const float*)d_in[4];
  const float* g1p  = (const float*)d_in[5];
  const float* be1  = (const float*)d_in[6];
  const float* g2p  = (const float*)d_in[7];
  const float* be2  = (const float*)d_in[8];
  const float* hwp  = (const float*)d_in[9];
  const float* hbp  = (const float*)d_in[10];
  float* outp = (float*)d_out;

  dim3 grid(B_SZ / MTILE);   // 64 workgroups, each owns 16 batch rows for all T
  dim3 block(THREADS);       // 16 wave32 waves
  hipLaunchKernelGGL(liquid_net_persistent_kernel, grid, block, 0, stream,
                     xp, Win, bin, taup, Wrec, g1p, be1, g2p, be2, hwp, hbp, outp);
}
